// WMLTextOneMultiHead_18408229830887
// MI455X (gfx1250) — compile-verified, matrix operand order checked by reading, not run on hardware
//
#include <hip/hip_runtime.h>
#include <hip/hip_bf16.h>

// ---------------------------------------------------------------------------
// Multi-head causal attention block for MI455X (gfx1250, wave32, WMMA).
//   B=8, T=1024, C=1024, H=16, D=64
// Pipeline:
//   0) one-time bf16 conversion/relayout of x and all weights
//   1) tile_gemm (qkv mode): xb @ Wb{q,k,v} -> q/k/v bf16 [B,H,T,D]
//   2) attn_flash: per (b,h, 64-row q block) online-softmax attention (WMMA)
//   3) tile_gemm (proj mode): att @ Wprojb + bias -> f32 out
// GEMM staging: double-buffered global_load_async_to_lds_b128 (GVS saddr form,
// ASYNCcnt partial waits) overlapping DMA with WMMA.
// ---------------------------------------------------------------------------

typedef __attribute__((ext_vector_type(16))) __bf16 v16bf;
typedef __attribute__((ext_vector_type(8)))  float  v8f;
typedef __attribute__((ext_vector_type(4)))  unsigned int u32x4;

union Frag16 {
  u32x4 u[2];
  v16bf bf;
};

union Pack8 {
  u32x4 v;
  unsigned short s[8];
};

__device__ __forceinline__ unsigned short f2bf(float f) {
  unsigned int u = __float_as_uint(f);
  u += 0x7FFFu + ((u >> 16) & 1u);   // round-to-nearest-even
  return (unsigned short)(u >> 16);
}

__device__ __forceinline__ v8f wmma_bf16(const Frag16& a, const Frag16& b, v8f c) {
  return __builtin_amdgcn_wmma_f32_16x16x32_bf16(false, a.bf, false, b.bf,
                                                 (short)0, c, false, false);
}

// Async DMA global->LDS, GVS mode: uniform 64-bit SGPR base + 32-bit VGPR off.
__device__ __forceinline__ void async_b128_s(unsigned lds_addr, unsigned voff,
                                             unsigned long long sbase) {
  asm volatile("global_load_async_to_lds_b128 %0, %1, %2"
               :: "v"(lds_addr), "v"(voff), "s"(sbase)
               : "memory");
}
__device__ __forceinline__ void wait_async0() {
  asm volatile("s_wait_asynccnt 0" ::: "memory");
}
__device__ __forceinline__ void wait_async8() {
  asm volatile("s_wait_asynccnt 8" ::: "memory");
}

#define TDIM 1024
#define CDIM 1024
#define HN   16
#define HD   64

// -------------------------------------------------------------------------
// Kernel 0a: x f32 -> bf16 (elementwise).  8 elems/thread.
// -------------------------------------------------------------------------
__global__ __launch_bounds__(256) void conv_x(const float* __restrict__ x,
                                              unsigned short* __restrict__ xb) {
  const size_t i = (size_t)(blockIdx.x * 256 + threadIdx.x) * 8;
  float4 a = *(const float4*)(x + i);
  float4 b = *(const float4*)(x + i + 4);
  Pack8 o;
  o.s[0] = f2bf(a.x); o.s[1] = f2bf(a.y); o.s[2] = f2bf(a.z); o.s[3] = f2bf(a.w);
  o.s[4] = f2bf(b.x); o.s[5] = f2bf(b.y); o.s[6] = f2bf(b.z); o.s[7] = f2bf(b.w);
  *(u32x4*)(xb + i) = o.v;
}

// -------------------------------------------------------------------------
// Kernel 0b: W{q,k,v}[h][c][d] f32 -> Wb[z][n][c] bf16 (n = h*64+d, K-contig)
// -------------------------------------------------------------------------
__global__ __launch_bounds__(256) void conv_wqkv(
    const float* __restrict__ Wq, const float* __restrict__ Wk,
    const float* __restrict__ Wv, unsigned short* __restrict__ Wb) {
  const int z = blockIdx.z;
  const float* W = (z == 0) ? Wq : ((z == 1) ? Wk : Wv);
  const size_t idx = (size_t)blockIdx.x * 256 + threadIdx.x;  // over 1M
  const int c = (int)(idx & 1023);
  const int n = (int)(idx >> 10);
  const float f = W[((size_t)(n >> 6) * CDIM + c) * HD + (n & 63)];
  Wb[(size_t)z * CDIM * CDIM + idx] = f2bf(f);
}

// -------------------------------------------------------------------------
// Kernel 0c: Wproj[k][n] f32 -> Wpb[n][k] bf16 (transpose, K-contig)
// -------------------------------------------------------------------------
__global__ __launch_bounds__(256) void conv_wproj(
    const float* __restrict__ Wp, unsigned short* __restrict__ Wpb) {
  const size_t idx = (size_t)blockIdx.x * 256 + threadIdx.x;  // over 1M
  const int k = (int)(idx & 1023);
  const int n = (int)(idx >> 10);
  Wpb[idx] = f2bf(Wp[(size_t)k * CDIM + n]);
}

// -------------------------------------------------------------------------
// Kernel 1/3: 128x128 block GEMM, 4 waves x (64x64), K-step 32,
// double-buffered async-LDS staging.
// -------------------------------------------------------------------------
#define LDS_TILE_BYTES (128 * 40 * 2)   // one A or B buffer

__global__ __launch_bounds__(128) void tile_gemm(
    const unsigned short* __restrict__ Abase,
    const unsigned short* __restrict__ Bbase,
    unsigned short* __restrict__ qkv_out,
    float* __restrict__ proj_out,
    const float* __restrict__ bias) {
  __shared__ unsigned short ldsA[2][128][40];  // [buf][m][k] (+8 pad)
  __shared__ unsigned short ldsB[2][128][40];  // [buf][n][k]

  const int tid  = threadIdx.x;
  const int lane = tid & 31;
  const int wave = tid >> 5;      // 0..3
  const int wr   = wave >> 1;     // 0..1 -> 64-row slab
  const int wc   = wave & 1;      // 0..1 -> 64-col slab
  const int bn   = blockIdx.x * 128;
  const int bm   = blockIdx.y * 128;
  const int z    = blockIdx.z;

  const unsigned short* A  = Abase;
  const unsigned short* Bm = Bbase + (size_t)z * CDIM * CDIM;

  const int hf = lane >> 4;
  const int ln = lane & 15;
  const int kb = hf * 8;

  v8f acc[4][4];
  #pragma unroll
  for (int i = 0; i < 4; ++i)
    #pragma unroll
    for (int j = 0; j < 4; ++j) acc[i][j] = (v8f)0.0f;

  // staging addresses: thread copies one 32-elem row of A and of B per tile
  const unsigned ldsa0 = (unsigned)(size_t)&ldsA[0][tid][0];
  const unsigned ldsb0 = (unsigned)(size_t)&ldsB[0][tid][0];
  const unsigned voff  = (unsigned)tid * (CDIM * 2);           // byte offset
  const unsigned long long sbaseA =
      (unsigned long long)(size_t)(A  + (size_t)bm * CDIM);
  const unsigned long long sbaseB =
      (unsigned long long)(size_t)(Bm + (size_t)bn * CDIM);

  // prologue: fill buffer 0 with k0 = 0
  #pragma unroll
  for (int i = 0; i < 4; ++i) {
    async_b128_s(ldsa0 + i * 16, voff + i * 16, sbaseA);
    async_b128_s(ldsb0 + i * 16, voff + i * 16, sbaseB);
  }

  for (int k0 = 0; k0 < CDIM; k0 += 32) {
    const int cur = (k0 >> 5) & 1;
    const bool more = (k0 + 32) < CDIM;
    if (more) {
      const unsigned la = ldsa0 + (cur ^ 1) * LDS_TILE_BYTES;
      const unsigned lb = ldsb0 + (cur ^ 1) * LDS_TILE_BYTES;
      const unsigned long long sa = sbaseA + (unsigned long long)(k0 + 32) * 2;
      const unsigned long long sb = sbaseB + (unsigned long long)(k0 + 32) * 2;
      #pragma unroll
      for (int i = 0; i < 4; ++i) {
        async_b128_s(la + i * 16, voff + i * 16, sa);
        async_b128_s(lb + i * 16, voff + i * 16, sb);
      }
      wait_async8();      // previous tile's 8 ops retired (in-order)
    } else {
      wait_async0();
    }
    __syncthreads();      // buffer `cur` ready for all waves

    Frag16 af[4], bf[4];
    #pragma unroll
    for (int mi = 0; mi < 4; ++mi) {
      const int m = wr * 64 + mi * 16 + ln;
      af[mi].u[0] = *(const u32x4*)&ldsA[cur][m][kb];
      af[mi].u[1] = *(const u32x4*)&ldsA[cur][m][kb + 16];
    }
    #pragma unroll
    for (int ni = 0; ni < 4; ++ni) {
      const int n = wc * 64 + ni * 16 + ln;
      bf[ni].u[0] = *(const u32x4*)&ldsB[cur][n][kb];
      bf[ni].u[1] = *(const u32x4*)&ldsB[cur][n][kb + 16];
    }
    #pragma unroll
    for (int mi = 0; mi < 4; ++mi)
      #pragma unroll
      for (int ni = 0; ni < 4; ++ni)
        acc[mi][ni] = wmma_bf16(af[mi], bf[ni], acc[mi][ni]);
    __syncthreads();      // all waves done reading `cur`
  }

  // ---- epilogue
  if (proj_out) {
    #pragma unroll
    for (int mi = 0; mi < 4; ++mi)
      #pragma unroll
      for (int ni = 0; ni < 4; ++ni)
        #pragma unroll
        for (int r = 0; r < 8; ++r) {
          const int row = bm + wr * 64 + mi * 16 + r + hf * 8;
          const int col = bn + wc * 64 + ni * 16 + ln;
          proj_out[(size_t)row * CDIM + col] = acc[mi][ni][r] + bias[col];
        }
  } else {
    unsigned short* outw = qkv_out + (size_t)z * (size_t)(8 * HN) * TDIM * HD;
    #pragma unroll
    for (int mi = 0; mi < 4; ++mi)
      #pragma unroll
      for (int ni = 0; ni < 4; ++ni)
        #pragma unroll
        for (int r = 0; r < 8; ++r) {
          const int row = bm + wr * 64 + mi * 16 + r + hf * 8;
          const int col = bn + wc * 64 + ni * 16 + ln;
          const int b = row >> 10, t = row & 1023;
          const int h = col >> 6,  d = col & 63;
          outw[(((size_t)b * HN + h) * TDIM + t) * HD + d] = f2bf(acc[mi][ni][r]);
        }
  }
}

// -------------------------------------------------------------------------
// Kernel 2: flash attention.  grid = (B*H, T/64), block = 128 (4 waves).
// -------------------------------------------------------------------------
__global__ __launch_bounds__(128) void attn_flash(
    const unsigned short* __restrict__ qw,
    const unsigned short* __restrict__ kw,
    const unsigned short* __restrict__ vw,
    unsigned short* __restrict__ ow) {
  __shared__ unsigned short ldsK[32][72];      // [s][d] 32x64 (+8 pad)
  __shared__ unsigned short ldsVT[64][40];     // [d][s] 64x32 (+8 pad)
  __shared__ unsigned short ldsP[4][16][40];   // per-wave P (16x32)

  const int tid  = threadIdx.x;
  const int lane = tid & 31;
  const int wave = tid >> 5;
  const int hf   = lane >> 4;
  const int ln   = lane & 15;
  const int kb   = hf * 8;

  const int bh   = blockIdx.x;
  const int mblk = blockIdx.y;
  const size_t base = (size_t)bh * TDIM * HD;
  const int t0 = mblk * 64 + wave * 16;

  Frag16 qf[2];
  {
    const unsigned short* qp = qw + base + (size_t)(t0 + ln) * HD;
    qf[0].u[0] = *(const u32x4*)(qp + kb);
    qf[0].u[1] = *(const u32x4*)(qp + kb + 16);
    qf[1].u[0] = *(const u32x4*)(qp + kb + 32);
    qf[1].u[1] = *(const u32x4*)(qp + kb + 48);
  }

  v8f accO[4];
  #pragma unroll
  for (int i = 0; i < 4; ++i) accO[i] = (v8f)0.0f;
  float m_i[8], l_i[8];
  #pragma unroll
  for (int j = 0; j < 8; ++j) { m_i[j] = -1e30f; l_i[j] = 0.0f; }

  const float scale = 0.125f;  // D^-0.5
  const int smax = (mblk + 1) * 64;

  // per-thread staging coords: thread covers 16 contiguous elems of the tile
  const int sv = tid >> 2;            // 0..31
  const int dv = (tid & 3) * 16;      // 0,16,32,48
  const unsigned ldskA = (unsigned)(size_t)&ldsK[sv][dv];
  const unsigned kvoff = (unsigned)(sv * (HD * 2) + dv * 2);   // bytes
  const unsigned long long kbase0 = (unsigned long long)(size_t)(kw + base);

  for (int s0 = 0; s0 < smax; s0 += 32) {
    // ---- K tile (32x64) via async DMA (GVS form)
    {
      const unsigned long long sb = kbase0 + (unsigned long long)s0 * (HD * 2);
      async_b128_s(ldskA,      kvoff,      sb);
      async_b128_s(ldskA + 16, kvoff + 16, sb);
    }
    // ---- V tile transposed: 2 vector loads + 16 scalar LDS stores
    {
      Pack8 p0, p1;
      const unsigned short* vp = vw + base + (size_t)(s0 + sv) * HD + dv;
      p0.v = *(const u32x4*)(vp);
      p1.v = *(const u32x4*)(vp + 8);
      #pragma unroll
      for (int i = 0; i < 8; ++i) {
        ldsVT[dv + i][sv]     = p0.s[i];
        ldsVT[dv + 8 + i][sv] = p1.s[i];
      }
    }
    wait_async0();
    __syncthreads();

    // ---- S(16x32) = Q @ K^T : two 16x16 C-tiles
    v8f S0 = (v8f)0.0f, S1 = (v8f)0.0f;
    {
      Frag16 kf;
      kf.u[0] = *(const u32x4*)&ldsK[ln][kb];
      kf.u[1] = *(const u32x4*)&ldsK[ln][kb + 16];
      S0 = wmma_bf16(qf[0], kf, S0);
      kf.u[0] = *(const u32x4*)&ldsK[ln][kb + 32];
      kf.u[1] = *(const u32x4*)&ldsK[ln][kb + 48];
      S0 = wmma_bf16(qf[1], kf, S0);
      kf.u[0] = *(const u32x4*)&ldsK[ln + 16][kb];
      kf.u[1] = *(const u32x4*)&ldsK[ln + 16][kb + 16];
      S1 = wmma_bf16(qf[0], kf, S1);
      kf.u[0] = *(const u32x4*)&ldsK[ln + 16][kb + 32];
      kf.u[1] = *(const u32x4*)&ldsK[ln + 16][kb + 48];
      S1 = wmma_bf16(qf[1], kf, S1);
    }

    // ---- online softmax (row r = VGPR j + 8*half)
    float alpha[8];
    const int c0 = s0 + ln, c1 = s0 + 16 + ln;
    #pragma unroll
    for (int j = 0; j < 8; ++j) {
      const int r = t0 + j + hf * 8;
      float v0 = S0[j] * scale; if (c0 > r) v0 = -1e30f;
      float v1 = S1[j] * scale; if (c1 > r) v1 = -1e30f;
      float mx = fmaxf(v0, v1);
      #pragma unroll
      for (int off = 1; off < 16; off <<= 1)
        mx = fmaxf(mx, __shfl_xor(mx, off, 32));
      const float mnew = fmaxf(m_i[j], mx);
      alpha[j] = __expf(m_i[j] - mnew);
      const float p0 = __expf(v0 - mnew);
      const float p1 = __expf(v1 - mnew);
      float rs = p0 + p1;
      #pragma unroll
      for (int off = 1; off < 16; off <<= 1)
        rs += __shfl_xor(rs, off, 32);
      l_i[j] = l_i[j] * alpha[j] + rs;
      m_i[j] = mnew;
      S0[j] = p0;
      S1[j] = p1;
    }

    // ---- C-layout -> A-fragment via per-wave LDS round trip
    #pragma unroll
    for (int j = 0; j < 8; ++j) {
      const int r = j + hf * 8;
      ldsP[wave][r][ln]      = f2bf(S0[j]);
      ldsP[wave][r][ln + 16] = f2bf(S1[j]);
    }
    __builtin_amdgcn_wave_barrier();
    asm volatile("s_wait_dscnt 0" ::: "memory");

    // ---- rescale O, O += P @ V
    #pragma unroll
    for (int dt = 0; dt < 4; ++dt)
      #pragma unroll
      for (int j = 0; j < 8; ++j) accO[dt][j] *= alpha[j];

    Frag16 pf;
    pf.u[0] = *(const u32x4*)&ldsP[wave][ln][kb];
    pf.u[1] = *(const u32x4*)&ldsP[wave][ln][kb + 16];
    #pragma unroll
    for (int dt = 0; dt < 4; ++dt) {
      Frag16 vf;
      const int d = dt * 16 + ln;
      vf.u[0] = *(const u32x4*)&ldsVT[d][kb];
      vf.u[1] = *(const u32x4*)&ldsVT[d][kb + 16];
      accO[dt] = wmma_bf16(pf, vf, accO[dt]);
    }
    __syncthreads();
  }

  // ---- epilogue: normalize, store bf16 to [B, T, H*D]
  const int b = bh >> 4, h = bh & 15;
  #pragma unroll
  for (int dt = 0; dt < 4; ++dt) {
    #pragma unroll
    for (int j = 0; j < 8; ++j) {
      const int r = t0 + j + hf * 8;
      const int d = dt * 16 + ln;
      const float v = accO[dt][j] / l_i[j];
      ow[((size_t)(b * TDIM + r)) * CDIM + h * HD + d] = f2bf(v);
    }
  }
}

// -------------------------------------------------------------------------
extern "C" void kernel_launch(void* const* d_in, const int* in_sizes, int n_in,
                              void* d_out, int out_size, void* d_ws, size_t ws_size,
                              hipStream_t stream) {
  const float* x     = (const float*)d_in[0];
  const float* Wq    = (const float*)d_in[1];
  const float* Wk    = (const float*)d_in[2];
  const float* Wv    = (const float*)d_in[3];
  const float* Wproj = (const float*)d_in[4];
  const float* bproj = (const float*)d_in[5];

  unsigned short* ws = (unsigned short*)d_ws;
  const size_t tsz = (size_t)8 * HN * TDIM * HD;    // 8,388,608 elems
  const size_t wsz = (size_t)CDIM * CDIM;           // 1,048,576 elems
  unsigned short* qw  = ws;
  unsigned short* kw  = ws + tsz;
  unsigned short* vw  = ws + 2 * tsz;
  unsigned short* owt = ws + 3 * tsz;
  unsigned short* xb  = ws + 4 * tsz;
  unsigned short* Wb  = ws + 5 * tsz;               // 3 matrices
  unsigned short* Wpb = ws + 5 * tsz + 3 * wsz;

  // 0) conversions / relayouts (one-time, bandwidth-trivial)
  conv_x<<<dim3((unsigned)(tsz / (256 * 8))), 256, 0, stream>>>(x, xb);
  conv_wqkv<<<dim3((unsigned)(wsz / 256), 1, 3), 256, 0, stream>>>(Wq, Wk, Wv, Wb);
  conv_wproj<<<dim3((unsigned)(wsz / 256)), 256, 0, stream>>>(Wproj, Wpb);

  // 1) QKV projections: M=8192, N=1024, K=1024, z = {q,k,v}
  tile_gemm<<<dim3(CDIM / 128, (8 * TDIM) / 128, 3), 128, 0, stream>>>(
      xb, Wb, qw, nullptr, nullptr);

  // 2) attention
  attn_flash<<<dim3(8 * HN, TDIM / 64), 128, 0, stream>>>(qw, kw, vw, owt);

  // 3) output projection + bias -> f32
  tile_gemm<<<dim3(CDIM / 128, (8 * TDIM) / 128, 1), 128, 0, stream>>>(
      owt, Wpb, nullptr, (float*)d_out, bproj);
}
